// GRUCov_72799695667429
// MI455X (gfx1250) — matrix operand order repaired and checked by reading, not compiled, not run on hardware
//
#include <hip/hip_runtime.h>

// GRU graph aggregator for MI455X (gfx1250, wave32, WMMA).
// f16 WMMA (16x16x32) with f32 accumulation; f32 recurrent state in registers.

#define D        128
#define KSTEPS   16
#define THREE_D  384
#define WS       392   // padded LDS stride (halves) for 384-wide transposed gate weights
#define SS       136   // padded LDS stride (halves) for 128-wide matrices / tiles
#define M_TILE   32
#define BLOCK    256

typedef _Float16 v16h  __attribute__((ext_vector_type(16)));
typedef _Float16 half8 __attribute__((ext_vector_type(8)));
typedef float    v8f   __attribute__((ext_vector_type(8)));
typedef float    f32x4 __attribute__((ext_vector_type(4)));

// LDS layout (in _Float16 elements)
#define OFF_WIH 0
#define OFF_WHH (128*WS)
#define OFF_WSE (2*128*WS)
#define OFF_WNE (2*128*WS + 128*SS)
#define OFF_H   (2*128*WS + 2*128*SS)
#define OFF_X0  (OFF_H  + 16*SS)
#define OFF_X1  (OFF_X0 + 16*SS)
#define SMEM_HALVES (OFF_X1 + 16*SS)   // 141696 halves = 283392 B < 320 KB

union AFrag { v16h v; half8 h[2]; };

// A fragment (16x32 f16, row-major tile in LDS): lane L<16 -> row L, K {kb..kb+7, kb+16..kb+23};
// lane L>=16 -> row L-16, K {kb+8.., kb+24..}. Two ds_load_b128 per lane.
__device__ __forceinline__ v16h load_a(const _Float16* buf, int stride, int kbase, int lane) {
  const int row = lane & 15;
  const int hi8 = (lane >> 4) << 3;
  const _Float16* p = buf + row * stride + kbase + hi8;
  AFrag f;
  f.h[0] = *(const half8*)(p);
  f.h[1] = *(const half8*)(p + 16);
  return f.v;
}

// B fragment (32x16 f16) from K-major transposed weights Wt[k][o]:
// lane L -> column obase+(L&15), K {kb+8h..kb+8h+7, kb+16+8h..kb+23+8h}, h = L>>4.
__device__ __forceinline__ v16h load_b(const _Float16* wt, int stride, int kbase, int obase, int lane) {
  const int hi8 = (lane >> 4) << 3;
  const _Float16* p = wt + (kbase + hi8) * stride + obase + (lane & 15);
  v16h b;
  #pragma unroll
  for (int s = 0; s < 8; ++s) {
    b[s]     = p[s * stride];
    b[8 + s] = p[(16 + s) * stride];
  }
  return b;
}

__device__ __forceinline__ float sigm_f(float x)  { return 1.0f / (1.0f + __expf(-x)); }
__device__ __forceinline__ float tanh_f(float x)  { return 2.0f / (1.0f + __expf(-2.0f * x)) - 1.0f; }

// convert one gathered/self row segment (8 f32 -> 8 f16) into an LDS tile
__device__ __forceinline__ void stage_seg(const float* __restrict__ src_row_seg,
                                          _Float16* __restrict__ dst_seg) {
  const f32x4* s = (const f32x4*)src_row_seg;
  f32x4 a = s[0], b = s[1];
  half8 h;
  h[0] = (_Float16)a[0]; h[1] = (_Float16)a[1]; h[2] = (_Float16)a[2]; h[3] = (_Float16)a[3];
  h[4] = (_Float16)b[0]; h[5] = (_Float16)b[1]; h[6] = (_Float16)b[2]; h[7] = (_Float16)b[3];
  *(half8*)dst_seg = h;
}

__global__ void __launch_bounds__(BLOCK)
gru_sage_wmma(const float* __restrict__ feat,
              const int*   __restrict__ nidx,
              const float* __restrict__ W_ih,  const float* __restrict__ W_hh,
              const float* __restrict__ b_ih,  const float* __restrict__ b_hh,
              const float* __restrict__ W_self,const float* __restrict__ W_neigh,
              const float* __restrict__ alpha,
              float* __restrict__ out, int n_nodes)
{
  extern __shared__ _Float16 smem[];
  _Float16* Wih  = smem + OFF_WIH;
  _Float16* Whh  = smem + OFF_WHH;
  _Float16* Wse  = smem + OFF_WSE;
  _Float16* Wne  = smem + OFF_WNE;
  _Float16* hbuf = smem + OFF_H;
  _Float16* xb0  = smem + OFF_X0;
  _Float16* xb1  = smem + OFF_X1;

  const int tid  = threadIdx.x;
  const int lane = tid & 31;
  const int wv   = tid >> 5;              // wave id: owns channels [16*wv, 16*wv+16)

  // ---- stage weights: global f32 row-major [o][k] -> LDS f16 K-major [k][o] ----
  for (int i = tid; i < THREE_D * D; i += BLOCK) {
    const int o = i / D, k = i - o * D;
    Wih[k * WS + o] = (_Float16)W_ih[i];
    Whh[k * WS + o] = (_Float16)W_hh[i];
  }
  for (int i = tid; i < D * D; i += BLOCK) {
    const int o = i / D, k = i - o * D;
    Wse[k * SS + o] = (_Float16)W_self[i];
    Wne[k * SS + o] = (_Float16)W_neigh[i];
  }

  const int ch  = (wv << 4) + (lane & 15);   // absolute hidden/output channel for this lane
  const float bi0 = b_ih[ch], bi1 = b_ih[D + ch], bi2 = b_ih[2 * D + ch];
  const float bh0 = b_hh[ch], bh1 = b_hh[D + ch], bh2 = b_hh[2 * D + ch];
  const float a_ch = alpha[ch];

  const int mrow = tid >> 4;      // staging: row within 16-row tile
  const int seg  = tid & 15;      // staging: 8-float segment within row
  const int obw  = wv << 4;

  for (int rt = 0; rt < M_TILE / 16; ++rt) {
    const int row_base = blockIdx.x * M_TILE + rt * 16;
    if (row_base >= n_nodes) break;

    // zero h exchange buffer (h0 = 0)
    for (int i = tid; i < 16 * SS; i += BLOCK) hbuf[i] = (_Float16)0.0f;

    // stage x(t=0): gather neighbor rows, f32 -> f16
    {
      int r = row_base + mrow; if (r >= n_nodes) r = n_nodes - 1;
      const int gidx = nidx[r * KSTEPS + 0];
      stage_seg(feat + (long)gidx * D + seg * 8, xb0 + mrow * SS + seg * 8);
    }

    float hprev[8];
    #pragma unroll
    for (int e = 0; e < 8; ++e) hprev[e] = 0.0f;

    for (int t = 0; t < KSTEPS; ++t) {
      __syncthreads();   // S1: xbuf[t&1] staged, hbuf stores from prev step visible

      const _Float16* xcur = (t & 1) ? xb1 : xb0;
      v16h xA[4], hA[4];
      #pragma unroll
      for (int kc = 0; kc < 4; ++kc) {
        xA[kc] = load_a(xcur, SS, kc * 32, lane);
        hA[kc] = load_a(hbuf, SS, kc * 32, lane);
      }

      // prefetch x(t+1) into the other buffer while WMMAs run
      if (t + 1 < KSTEPS) {
        int r = row_base + mrow; if (r >= n_nodes) r = n_nodes - 1;
        const int gidx = nidx[r * KSTEPS + (t + 1)];
        _Float16* nxt = ((t + 1) & 1) ? xb1 : xb0;
        stage_seg(feat + (long)gidx * D + seg * 8, nxt + mrow * SS + seg * 8);
      }

      // accumulators seeded with biases (uniform per lane-column)
      v8f gi[3], gh[3];
      #pragma unroll
      for (int e = 0; e < 8; ++e) {
        gi[0][e] = bi0; gi[1][e] = bi1; gi[2][e] = bi2;
        gh[0][e] = bh0; gh[1][e] = bh1; gh[2][e] = bh2;
      }

      #pragma unroll
      for (int g = 0; g < 3; ++g) {
        const int obase = g * D + obw;
        #pragma unroll
        for (int kc = 0; kc < 4; ++kc) {
          v16h bwi = load_b(Wih, WS, kc * 32, obase, lane);
          gi[g] = __builtin_amdgcn_wmma_f32_16x16x32_f16(false, xA[kc], false, bwi,
                                                         (short)0, gi[g], false, false);
          v16h bwh = load_b(Whh, WS, kc * 32, obase, lane);
          gh[g] = __builtin_amdgcn_wmma_f32_16x16x32_f16(false, hA[kc], false, bwh,
                                                         (short)0, gh[g], false, false);
        }
      }

      // GRU gate math (wave-local: r,z,n tiles are this wave's own channel slice)
      #pragma unroll
      for (int e = 0; e < 8; ++e) {
        const float r = sigm_f(gi[0][e] + gh[0][e]);
        const float z = sigm_f(gi[1][e] + gh[1][e]);
        const float n = tanh_f(gi[2][e] + r * gh[2][e]);
        hprev[e] = (1.0f - z) * n + z * hprev[e];
      }

      __syncthreads();   // S2: everyone finished reading hbuf before we overwrite our slice

      {
        const int hi8 = (lane >> 4) << 3;
        #pragma unroll
        for (int e = 0; e < 8; ++e)
          hbuf[(e + hi8) * SS + ch] = (_Float16)hprev[e];
      }
    }

    // ---- output: feat @ W_self^T + h @ W_neigh^T, PReLU ----
    {
      int r = row_base + mrow; if (r >= n_nodes) r = n_nodes - 1;
      stage_seg(feat + (long)r * D + seg * 8, xb0 + mrow * SS + seg * 8);
    }
    __syncthreads();   // final h + self-feat tile visible

    v16h fA[4], hA2[4];
    #pragma unroll
    for (int kc = 0; kc < 4; ++kc) {
      fA[kc]  = load_a(xb0,  SS, kc * 32, lane);
      hA2[kc] = load_a(hbuf, SS, kc * 32, lane);
    }
    v8f acc = {0.f, 0.f, 0.f, 0.f, 0.f, 0.f, 0.f, 0.f};
    #pragma unroll
    for (int kc = 0; kc < 4; ++kc) {
      v16h bse = load_b(Wse, SS, kc * 32, obw, lane);
      acc = __builtin_amdgcn_wmma_f32_16x16x32_f16(false, fA[kc], false, bse,
                                                   (short)0, acc, false, false);
    }
    #pragma unroll
    for (int kc = 0; kc < 4; ++kc) {
      v16h bne = load_b(Wne, SS, kc * 32, obw, lane);
      acc = __builtin_amdgcn_wmma_f32_16x16x32_f16(false, hA2[kc], false, bne,
                                                   (short)0, acc, false, false);
    }

    const int hi8 = (lane >> 4) << 3;
    #pragma unroll
    for (int e = 0; e < 8; ++e) {
      const int row = row_base + e + hi8;
      float v = acc[e];
      v = (v >= 0.0f) ? v : a_ch * v;
      if (row < n_nodes) out[(long)row * D + ch] = v;
    }
    __syncthreads();   // before next rt re-zeros hbuf / restages xb0
  }
}

extern "C" void kernel_launch(void* const* d_in, const int* in_sizes, int n_in,
                              void* d_out, int out_size, void* d_ws, size_t ws_size,
                              hipStream_t stream) {
  const float* feat    = (const float*)d_in[0];
  const int*   nidx    = (const int*)  d_in[1];
  const float* W_ih    = (const float*)d_in[2];
  const float* W_hh    = (const float*)d_in[3];
  const float* b_ih    = (const float*)d_in[4];
  const float* b_hh    = (const float*)d_in[5];
  const float* W_self  = (const float*)d_in[6];
  const float* W_neigh = (const float*)d_in[7];
  const float* alpha   = (const float*)d_in[8];

  const int n_nodes = in_sizes[0] / D;                  // 50000
  const int blocks  = (n_nodes + M_TILE - 1) / M_TILE;  // 1563
  const size_t shmem = SMEM_HALVES * sizeof(_Float16);  // ~283 KB (< 320 KB WGP LDS)

  gru_sage_wmma<<<blocks, BLOCK, shmem, stream>>>(
      feat, nidx, W_ih, W_hh, b_ih, b_hh, W_self, W_neigh, alpha,
      (float*)d_out, n_nodes);
}
